// MGRU_59261958750752
// MI455X (gfx1250) — matrix operand order repaired and checked by reading, not compiled, not run on hardware
//
#include <hip/hip_runtime.h>

#define TT 2048
#define BB 64
#define DD 128
#define HH 128
#define HSTR 136   // LDS row stride in halves (272B = 17*16B): 16B-aligned, rotates banks by 4/row

typedef __attribute__((ext_vector_type(16))) _Float16 v16h;
typedef __attribute__((ext_vector_type(8)))  float    v8f;

union AFrag { v16h v; uint4 q[2]; };
union BFrag { v16h v; _Float16 h[16]; };
union HPack { _Float16 h[8]; uint4 q; };

// Split workgroup barrier (CDNA5): producer side waits out its own LDS ops, then signals.
// Independent work can be placed between signal and wait.
__device__ __forceinline__ void wg_barrier_signal() {
  asm volatile("s_wait_dscnt 0x0\n\ts_barrier_signal -1" ::: "memory");
}
__device__ __forceinline__ void wg_barrier_wait() {
  asm volatile("s_barrier_wait -1" ::: "memory");
}

// A-matrix 16x32 f16 fragment from an LDS f16 tile [16][HSTR].
// ISA layout: lanes 0-15 (M=lane): V0-3 = K 0..7, V4-7 = K 16..23; lanes 16-31: +8.
__device__ __forceinline__ v16h load_afrag(const _Float16* buf, int kbase, int lane) {
  const int m  = lane & 15;
  const int kh = (lane & 16) ? 8 : 0;
  const _Float16* p = buf + m * HSTR + kbase + kh;
  AFrag u;
  u.q[0] = *(const uint4*)(p);        // ds_load_b128: K kbase+kh .. +7
  u.q[1] = *(const uint4*)(p + 16);   // ds_load_b128: K kbase+kh+16 .. +23
  return u.v;
}

// B-matrix 32x16 f16 fragment from row-major f32 matrix M[ld]: B[k][n] = M[colbase+n][kstep*32+k].
// Layout (matches A): lanes 0-15 hold K 0..7,16..23 (col=lane), lanes 16-31 hold K 8..15,24..31.
__device__ __forceinline__ v16h make_bfrag(const float* M, int ld, int colbase, int kstep, int lane) {
  const int n  = lane & 15;
  const int kh = (lane & 16) ? 8 : 0;
  const float* src = M + (colbase + n) * ld + kstep * 32 + kh;
  BFrag u;
#pragma unroll
  for (int i = 0; i < 16; ++i) u.h[i] = (_Float16)src[i < 8 ? i : i + 8];
  return u.v;
}

// Hardware tanh (CDNA5 TRANS op v_tanh_f32, confirmed in disasm).
__device__ __forceinline__ float vtanh(float x) {
#if __has_builtin(__builtin_amdgcn_tanhf)
  return __builtin_amdgcn_tanhf(x);
#else
  return 1.0f - 2.0f * __builtin_amdgcn_rcpf(1.0f + __builtin_amdgcn_exp2f(2.885390082f * x));
#endif
}

__global__ __launch_bounds__(256, 1)
void mgru_scan_kernel(const float* __restrict__ x,
                      const float* __restrict__ W_w, const float* __restrict__ W_b,
                      const float* __restrict__ Uf_w, const float* __restrict__ Uf_b,
                      const float* __restrict__ Uh_w, const float* __restrict__ Uh_b,
                      float* __restrict__ out)
{
  __shared__ __attribute__((aligned(16))) _Float16 h_buf[16 * HSTR];
  __shared__ __attribute__((aligned(16))) _Float16 a_buf[16 * HSTR];
  __shared__ __attribute__((aligned(16))) _Float16 x_buf[2][16 * HSTR];

  const int g    = blockIdx.x;          // batch row-tile (16 rows each), 4 blocks
  const int tid  = threadIdx.x;
  const int w    = tid >> 5;            // wave id 0..7 -> output column tile
  const int lane = tid & 31;
  const int n    = lane & 15;
  const int mb   = (lane & 16) ? 8 : 0; // C/D layout: VGPR r <-> row mb+r, col 16w+n
  const int jcol = 16 * w + n;

  // Staging geometry: 256 threads x 8 elements = one 16x128 tile
  const int srow = (tid * 8) >> 7;
  const int scol = (tid * 8) & 127;

  // Resident f16 B-fragments (converted once from f32 weights)
  v16h BUf[4], BUh[4], BWf[4], BWh[4];
#pragma unroll
  for (int s = 0; s < 4; ++s) {
    BUf[s] = make_bfrag(Uf_w, HH, 16 * w,      s, lane);
    BUh[s] = make_bfrag(Uh_w, HH, 16 * w,      s, lane);
    BWf[s] = make_bfrag(W_w,  DD, 16 * w,      s, lane);
    BWh[s] = make_bfrag(W_w,  DD, HH + 16 * w, s, lane);
  }
  const float bUfh = 0.5f * Uf_b[jcol];   // pre-halved for sigmoid-via-tanh
  const float bUh  = Uh_b[jcol];
  const float bWf  = W_b[jcol];
  const float bWh  = W_b[HH + jcol];

  // h0 = 0 (register copy in C/D layout + f16 mirror in LDS)
#pragma unroll
  for (int i = 0; i < 8; ++i) h_buf[srow * HSTR + scol + i] = (_Float16)0.0f;
  float hreg[8];
#pragma unroll
  for (int r = 0; r < 8; ++r) hreg[r] = 0.0f;

  // Stage x[0], x[1] into LDS (f32 -> f16)
#pragma unroll
  for (int tt = 0; tt < 2; ++tt) {
    const float* xs = x + (size_t)tt * (BB * DD) + (size_t)(16 * g + srow) * DD + scol;
    HPack xp;
#pragma unroll
    for (int i = 0; i < 8; ++i) xp.h[i] = (_Float16)xs[i];
    *(uint4*)&x_buf[tt][srow * HSTR + scol] = xp.q;
  }
  __syncthreads();

  // Pre-compute xw for t=0 from x_buf[0]
  v8f xwf, xwh;
#pragma unroll
  for (int r = 0; r < 8; ++r) { xwf[r] = bWf; xwh[r] = bWh; }
#pragma unroll
  for (int s = 0; s < 4; ++s) {
    v16h a = load_afrag(x_buf[0], 32 * s, lane);
    xwf = __builtin_amdgcn_wmma_f32_16x16x32_f16(false, a, false, BWf[s], (short)0, xwf, false, false);
    xwh = __builtin_amdgcn_wmma_f32_16x16x32_f16(false, a, false, BWh[s], (short)0, xwh, false, false);
  }
  __syncthreads();   // protect x_buf[0] from step-0 restaging

  const size_t outHS = (size_t)BB * HH;

  for (int t = 0; t < TT; ++t) {
    // ---- issue x[t+2] global loads early (independent; latency overlaps GEMM1 + barrier) ----
    float xstage[8];
    if (t + 2 < TT) {
      const float* xs = x + (size_t)(t + 2) * (BB * DD) + (size_t)(16 * g + srow) * DD + scol;
#pragma unroll
      for (int i = 0; i < 8; ++i) xstage[i] = xs[i];
    }

    // ---- GEMM1: f_pre = h @ Uf^T + xw_f ----
    v8f facc = xwf;
#pragma unroll
    for (int s = 0; s < 4; ++s) {
      v16h a = load_afrag(h_buf, 32 * s, lane);
      facc = __builtin_amdgcn_wmma_f32_16x16x32_f16(false, a, false, BUf[s], (short)0, facc, false, false);
    }

    // Independent f32->f16 x-stage conversion here: fills the 4 co-exec
    // hazard slots between the last GEMM1 WMMA and the dependent FMAs below.
    HPack xp;
    if (t + 2 < TT) {
#pragma unroll
      for (int i = 0; i < 8; ++i) xp.h[i] = (_Float16)xstage[i];
    }

    // sigmoid(z) = 0.5 + 0.5*tanh(0.5*z)
    float f[8];
#pragma unroll
    for (int r = 0; r < 8; ++r)
      f[r] = fmaf(0.5f, vtanh(fmaf(0.5f, facc[r], bUfh)), 0.5f);

    // a = f * h -> LDS (f16)
#pragma unroll
    for (int r = 0; r < 8; ++r)
      a_buf[(mb + r) * HSTR + jcol] = (_Float16)(f[r] * hreg[r]);

    wg_barrier_signal();   // s_wait_dscnt 0 + s_barrier_signal -1

    // ===== overlap window: independent work while other waves arrive =====
    // GEMM0: xw for step t+1 (reads x_buf[(t+1)&1], untouched this phase)
    if (t + 1 < TT) {
      v8f nf, nh;
#pragma unroll
      for (int r = 0; r < 8; ++r) { nf[r] = bWf; nh[r] = bWh; }
#pragma unroll
      for (int s = 0; s < 4; ++s) {
        v16h a = load_afrag(x_buf[(t + 1) & 1], 32 * s, lane);
        nf = __builtin_amdgcn_wmma_f32_16x16x32_f16(false, a, false, BWf[s], (short)0, nf, false, false);
        nh = __builtin_amdgcn_wmma_f32_16x16x32_f16(false, a, false, BWh[s], (short)0, nh, false, false);
      }
      xwf = nf; xwh = nh;
    }
    // single ds_store_b128 of the staged tile slice into the freed buffer
    if (t + 2 < TT)
      *(uint4*)&x_buf[t & 1][srow * HSTR + scol] = xp.q;
    // =====================================================================

    wg_barrier_wait();

    // ---- GEMM2: hh_pre = (f*h) @ Uh^T + xw_h ----
    v8f hacc = xwh;
#pragma unroll
    for (int s = 0; s < 4; ++s) {
      v16h a = load_afrag(a_buf, 32 * s, lane);
      hacc = __builtin_amdgcn_wmma_f32_16x16x32_f16(false, a, false, BUh[s], (short)0, hacc, false, false);
    }

    // ---- update: h = h + f*(tanh(hh_pre) - h); f16 mirror to LDS ----
    float hn[8];
#pragma unroll
    for (int r = 0; r < 8; ++r) {
      const float hh = vtanh(hacc[r] + bUh);
      hn[r] = fmaf(f[r], hh - hreg[r], hreg[r]);
      hreg[r] = hn[r];
      h_buf[(mb + r) * HSTR + jcol] = (_Float16)hn[r];
    }

    wg_barrier_signal();   // h_buf published

    // global hs[t] stores slide into the end-barrier window (streamed, non-temporal)
    float* outp = out + (size_t)t * outHS + (size_t)(16 * g + mb) * HH + jcol;
#pragma unroll
    for (int r = 0; r < 8; ++r)
      __builtin_nontemporal_store(hn[r], outp + (size_t)r * HH);
    if (t == TT - 1) {
      float* lastp = out + (size_t)TT * outHS + (size_t)(16 * g + mb) * HH + jcol;
#pragma unroll
      for (int r = 0; r < 8; ++r)
        __builtin_nontemporal_store(hreg[r], lastp + (size_t)r * HH);
    }

    wg_barrier_wait();
  }
}

extern "C" void kernel_launch(void* const* d_in, const int* in_sizes, int n_in,
                              void* d_out, int out_size, void* d_ws, size_t ws_size,
                              hipStream_t stream) {
  (void)in_sizes; (void)n_in; (void)d_ws; (void)ws_size; (void)out_size;
  const float* x    = (const float*)d_in[0];
  const float* W_w  = (const float*)d_in[1];
  const float* W_b  = (const float*)d_in[2];
  const float* Uf_w = (const float*)d_in[3];
  const float* Uf_b = (const float*)d_in[4];
  const float* Uh_w = (const float*)d_in[5];
  const float* Uh_b = (const float*)d_in[6];
  float* out = (float*)d_out;

  dim3 grid(BB / 16);   // 4 workgroups, one 16-row batch tile each
  dim3 block(256);      // 8 wave32 -> 8 column tiles of H=128
  hipLaunchKernelGGL(mgru_scan_kernel, grid, block, 0, stream,
                     x, W_w, W_b, Uf_w, Uf_b, Uh_w, Uh_b, out);
}
